// HybridAttention_55052890800495
// MI455X (gfx1250) — matrix operand order repaired and checked by reading, not compiled, not run on hardware
//
#include <hip/hip_runtime.h>

// ---------------------------------------------------------------------------
// HybridAttention: out = (softmax(s*QK^T) + softmax_local(s*QK^T)) @ V
// B=4, L=2048, H=8, E=64, layout [B, L, H, E], fp32 in/out, scale = 1/8.
// Kernel 1: fused flash attention (global branch) with bf16 WMMA, computed
//           TRANSPOSED (S^T = K Q^T, O^T = V^T P^T) so that softmax row stats
//           are per-lane scalars and P staging is packed ds_store_b128.
//           scale (=2^-3, exact) is folded into the Q fragments.
// Kernel 2: 4-wide local-window branch, accumulated into the output.
// ---------------------------------------------------------------------------

typedef __bf16 bf16;
typedef bf16  bf16x8  __attribute__((ext_vector_type(8)));
typedef bf16  bf16x16 __attribute__((ext_vector_type(16)));
typedef float floatx4 __attribute__((ext_vector_type(4)));
typedef float floatx8 __attribute__((ext_vector_type(8)));

union Frag16 {
    bf16x16 v;
    bf16x8  h[2];
};

#define WMMA_BF16(A, B, C) \
    __builtin_amdgcn_wmma_f32_16x16x32_bf16(false, (A), false, (B), (short)0, (C), false, false)

#define BQ 4
#define LQ 2048
#define HQ 8
#define EQ 64
#define SCALE 0.125f
#define ROWSTRIDE (HQ * EQ)   // 512 floats between consecutive l for fixed (b,h)
#define NTILES (LQ / 64)      // 32 key tiles

// ---------------------------------------------------------------------------
// Kernel 1: global (unmasked) flash attention. Writes out = A_global @ V.
// Block: 128 threads = 4 waves. Each wave: 16 query rows (the N/column dim of
// the transposed fragments). Block: 64 rows of one (b,h).
// ---------------------------------------------------------------------------
__global__ __launch_bounds__(128) void hybrid_attn_global(
    const float* __restrict__ Q, const float* __restrict__ K,
    const float* __restrict__ V, float* __restrict__ Out)
{
    __shared__ bf16 kT[64 * 64];        // K tile  [n][e]
    __shared__ bf16 vT[64 * 64];        // V tile  [e][n]  (transposed)
    __shared__ bf16 pS[4][16 * 64];     // per-wave P staging [m][n]

    const int blk   = blockIdx.x;       // 0..1023
    const int mtile = blk & 31;         // L/64 query tiles
    const int bh    = blk >> 5;
    const int h     = bh & (HQ - 1);
    const int b     = bh >> 3;

    const int tid  = threadIdx.x;
    const int wave = tid >> 5;
    const int lane = tid & 31;
    const int lg   = lane >> 4;         // half-wave group: 0 or 1
    const int ln   = lane & 15;

    const size_t base = ((size_t)b * LQ * HQ + h) * EQ;   // + l*512 + e
    const int m0 = mtile * 64 + wave * 16;                // first query row

    // ---- Load (scale*Q) strip as two B fragments of Q^T (col m = ln) ----
    // Fragment ke covers e in [32ke, 32ke+32); element j <-> e = 32ke+lg*8+j (j<8), 32ke+16+lg*8+(j-8) (j>=8).
    // SCALE = 2^-3 is exact through the bf16 convert, so S^T needs no scaling.
    Frag16 bq[2];
    {
        const float* qrow = Q + base + (size_t)(m0 + ln) * ROWSTRIDE;
        #pragma unroll
        for (int ke = 0; ke < 2; ++ke) {
            const float* qp = qrow + ke * 32 + lg * 8;
            floatx4 q0 = *(const floatx4*)(qp + 0);
            floatx4 q1 = *(const floatx4*)(qp + 4);
            floatx4 q2 = *(const floatx4*)(qp + 16);
            floatx4 q3 = *(const floatx4*)(qp + 20);
            #pragma unroll
            for (int i = 0; i < 4; ++i) {
                bq[ke].h[0][i]     = (bf16)(q0[i] * SCALE);
                bq[ke].h[0][i + 4] = (bf16)(q1[i] * SCALE);
                bq[ke].h[1][i]     = (bf16)(q2[i] * SCALE);
                bq[ke].h[1][i + 4] = (bf16)(q3[i] * SCALE);
            }
        }
    }

    // ---- O^T accumulators: o[et][r] = O[m=ln][e = et*16 + lg*8 + r] ----
    floatx8 o[4];
    #pragma unroll
    for (int et = 0; et < 4; ++et) o[et] = (floatx8){0,0,0,0,0,0,0,0};
    // Per-lane scalar softmax state for query row m = m0 + ln.
    float m_run = -1.0e30f;
    float s_run = 0.0f;

    // ---- Stream over 32 key tiles of 64 ----
    for (int t = 0; t < NTILES; ++t) {
        const int s0 = t * 64;

        // Cooperative staging: K -> kT[n][e], V -> vT[e][n], both bf16.
        #pragma unroll
        for (int it = 0; it < 8; ++it) {
            const int f4 = it * 128 + tid;        // 1024 float4 per tile
            const int n  = f4 >> 4;
            const int e4 = (f4 & 15) << 2;
            const float* src = K + base + (size_t)(s0 + n) * ROWSTRIDE + e4;
            const float* srv = V + base + (size_t)(s0 + n) * ROWSTRIDE + e4;
            floatx4 kv = *(const floatx4*)src;
            floatx4 vv = *(const floatx4*)srv;
            #pragma unroll
            for (int j = 0; j < 4; ++j) {
                kT[n * 64 + e4 + j]   = (bf16)kv[j];
                vT[(e4 + j) * 64 + n] = (bf16)vv[j];
            }
        }

        // Prefetch next tile into cache (global_prefetch_b8), uniform branch.
        if (t + 1 < NTILES) {
            const size_t pofs = base + (size_t)(s0 + 64 + (tid >> 1)) * ROWSTRIDE
                              + (tid & 1) * 32;
            __builtin_prefetch(K + pofs, 0, 3);
            __builtin_prefetch(V + pofs, 0, 3);
        }
        __syncthreads();

        // ---- S^T = (scale*Q K^T)^T : st[nt][r], key n = nt*16+lg*8+r ----
        floatx8 st[4];
        #pragma unroll
        for (int nt = 0; nt < 4; ++nt) {
            const bf16* kp = &kT[(nt * 16 + ln) * 64 + lg * 8];
            Frag16 ak0, ak1;
            ak0.h[0] = *(const bf16x8*)(kp + 0);   // e = lg*8 .. +7
            ak0.h[1] = *(const bf16x8*)(kp + 16);  // e = 16+lg*8 .. +7
            ak1.h[0] = *(const bf16x8*)(kp + 32);
            ak1.h[1] = *(const bf16x8*)(kp + 48);
            floatx8 c = (floatx8){0,0,0,0,0,0,0,0};
            c = WMMA_BF16(ak0.v, bq[0].v, c);
            c = WMMA_BF16(ak1.v, bq[1].v, c);
            st[nt] = c;
        }

        // ---- Column (= query-row) max: 31 in-lane ops + one cross-half shfl
        float tmax = st[0][0];
        #pragma unroll
        for (int nt = 0; nt < 4; ++nt)
            #pragma unroll
            for (int r = 0; r < 8; ++r)
                tmax = fmaxf(tmax, st[nt][r]);
        tmax = fmaxf(tmax, __shfl_xor(tmax, 16, 32));

        // ---- Online rescale (all per-lane scalars) ----
        const float mnew  = fmaxf(m_run, tmax);
        const float alpha = __expf(m_run - mnew);
        m_run = mnew;
        s_run *= alpha;
        #pragma unroll
        for (int et = 0; et < 4; ++et)
            #pragma unroll
            for (int r = 0; r < 8; ++r)
                o[et][r] *= alpha;

        // ---- P = exp(S - m); packed bf16x8 stores (one ds_store_b128 / nt)
        float rsum = 0.0f;
        #pragma unroll
        for (int nt = 0; nt < 4; ++nt) {
            bf16x8 pv;
            #pragma unroll
            for (int r = 0; r < 8; ++r) {
                float p = __expf(st[nt][r] - mnew);
                rsum += p;
                pv[r] = (bf16)p;
            }
            *(bf16x8*)&pS[wave][ln * 64 + nt * 16 + lg * 8] = pv;
        }
        rsum += __shfl_xor(rsum, 16, 32);
        s_run += rsum;

        // ---- Reload P^T as B fragments (lane reads its own pS row) ----
        Frag16 bp[2];
        #pragma unroll
        for (int ke = 0; ke < 2; ++ke) {
            const bf16* pp = &pS[wave][ln * 64 + ke * 32 + lg * 8];
            bp[ke].h[0] = *(const bf16x8*)(pp + 0);
            bp[ke].h[1] = *(const bf16x8*)(pp + 16);
        }

        // ---- O^T += V^T P^T ----
        #pragma unroll
        for (int et = 0; et < 4; ++et) {
            const bf16* vp = &vT[(et * 16 + ln) * 64 + lg * 8];
            Frag16 av0, av1;
            av0.h[0] = *(const bf16x8*)(vp + 0);   // n = lg*8 .. +7
            av0.h[1] = *(const bf16x8*)(vp + 16);  // n = 16+lg*8 .. +7
            av1.h[0] = *(const bf16x8*)(vp + 32);
            av1.h[1] = *(const bf16x8*)(vp + 48);
            o[et] = WMMA_BF16(av0.v, bp[0].v, o[et]);
            o[et] = WMMA_BF16(av1.v, bp[1].v, o[et]);
        }
        __syncthreads();
    }

    // ---- Normalize and store: lane owns query row m0+ln, 8-float chunks ----
    const float inv = 1.0f / s_run;
    float* op = Out + base + (size_t)(m0 + ln) * ROWSTRIDE;
    #pragma unroll
    for (int et = 0; et < 4; ++et) {
        floatx4 lo = { o[et][0] * inv, o[et][1] * inv, o[et][2] * inv, o[et][3] * inv };
        floatx4 hi = { o[et][4] * inv, o[et][5] * inv, o[et][6] * inv, o[et][7] * inv };
        *(floatx4*)(op + et * 16 + lg * 8 + 0) = lo;
        *(floatx4*)(op + et * 16 + lg * 8 + 4) = hi;
    }
}

// ---------------------------------------------------------------------------
// Kernel 2: local windowed branch: j in [max(l-2,0), min(l+2,L)).
// One wave per (b,l,h) row; each lane owns 2 of the 64 output elements.
// Accumulates into Out (each element touched by exactly one lane).
// ---------------------------------------------------------------------------
__global__ __launch_bounds__(128) void hybrid_attn_local(
    const float* __restrict__ Q, const float* __restrict__ K,
    const float* __restrict__ V, float* __restrict__ Out)
{
    const int gwave = (int)((blockIdx.x * 128 + threadIdx.x) >> 5); // (b*L+l)*H+h
    const int lane  = threadIdx.x & 31;
    const int l     = (gwave >> 3) & (LQ - 1);

    const ptrdiff_t rowQ = (ptrdiff_t)gwave * EQ;
    const float* qp = Q + rowQ + 2 * lane;
    const float q0 = qp[0], q1 = qp[1];

    float s[4];
    bool  valid[4];
    #pragma unroll
    for (int d = 0; d < 4; ++d) {
        const int j = l + d - 2;
        valid[d] = (j >= 0) && (j < LQ);
        float part = 0.0f;
        if (valid[d]) {
            const float* kp = K + rowQ + (ptrdiff_t)(d - 2) * ROWSTRIDE + 2 * lane;
            part = q0 * kp[0] + q1 * kp[1];
        }
        #pragma unroll
        for (int msk = 1; msk < 32; msk <<= 1)
            part += __shfl_xor(part, msk, 32);
        s[d] = part * SCALE;
    }

    float mx = -1.0e30f;
    #pragma unroll
    for (int d = 0; d < 4; ++d)
        if (valid[d]) mx = fmaxf(mx, s[d]);

    float p[4], den = 0.0f;
    #pragma unroll
    for (int d = 0; d < 4; ++d) {
        p[d] = valid[d] ? __expf(s[d] - mx) : 0.0f;
        den += p[d];
    }
    const float inv = 1.0f / den;

    float a0 = 0.0f, a1 = 0.0f;
    #pragma unroll
    for (int d = 0; d < 4; ++d) {
        if (valid[d]) {
            const float* vp = V + rowQ + (ptrdiff_t)(d - 2) * ROWSTRIDE + 2 * lane;
            const float w = p[d] * inv;
            a0 += w * vp[0];
            a1 += w * vp[1];
        }
    }

    float* op = Out + rowQ + 2 * lane;
    op[0] += a0;
    op[1] += a1;
}

// ---------------------------------------------------------------------------
extern "C" void kernel_launch(void* const* d_in, const int* in_sizes, int n_in,
                              void* d_out, int out_size, void* d_ws, size_t ws_size,
                              hipStream_t stream)
{
    (void)in_sizes; (void)n_in; (void)out_size; (void)d_ws; (void)ws_size;
    const float* Q = (const float*)d_in[0];
    const float* K = (const float*)d_in[1];
    const float* V = (const float*)d_in[2];
    float* Out = (float*)d_out;

    // Kernel 1: B*H*(L/64) blocks, 4 waves each (global branch, overwrites Out).
    hipLaunchKernelGGL(hybrid_attn_global,
                       dim3(BQ * HQ * (LQ / 64)), dim3(128), 0, stream,
                       Q, K, V, Out);

    // Kernel 2: one wave per (b,l,h) row (local branch, accumulates into Out).
    hipLaunchKernelGGL(hybrid_attn_local,
                       dim3((BQ * LQ * HQ) / 4), dim3(128), 0, stream,
                       Q, K, V, Out);
}
